// SG_GNN_8521215115885
// MI455X (gfx1250) — compile-verified, matrix operand order checked by reading, not run on hardware
//
#include <hip/hip_runtime.h>
#include <hip/hip_bf16.h>
#include <math.h>

// ---------------------------------------------------------------------------
// MI455X (gfx1250) GNN forward: 2x (SAGE + GAT) with SiLU-gated residual.
// Edge passes  = L2-resident float atomics (scatter targets << 192MB L2).
// Dense GEMMs  = v_wmma_f32_16x16x32_bf16, fp32 accumulate, wave32 tiles,
//                weights pre-packed once into WMMA B-fragment layout.
// ---------------------------------------------------------------------------

#define TPB 256

typedef __attribute__((ext_vector_type(16))) __bf16 v16bf;
typedef __attribute__((ext_vector_type(8)))  float  v8f;

static inline dim3 gridFor(size_t n) { return dim3((unsigned)((n + TPB - 1) / TPB)); }

// ------------------------------ utility kernels ----------------------------

__global__ void init_f32(float* __restrict__ p, float v, size_t n) {
  size_t i = (size_t)blockIdx.x * blockDim.x + threadIdx.x;
  if (i < n) p[i] = v;
}

__global__ void deg_kernel(const int* __restrict__ dst, float* __restrict__ deg, int E) {
  int e = blockIdx.x * blockDim.x + threadIdx.x;
  if (e < E) atomicAdd(&deg[dst[e]], 1.0f);
}

// agg[dst] += x[src], 4 floats per thread (16 threads per edge)
__global__ void sage_agg(const float* __restrict__ x, const int* __restrict__ src,
                         const int* __restrict__ dst, float* __restrict__ agg, int E) {
  size_t t = (size_t)blockIdx.x * blockDim.x + threadIdx.x;
  int e = (int)(t >> 4);
  if (e >= E) return;
  int g = (int)(t & 15);
  int s = src[e], d = dst[e];
  const float4 v = *(const float4*)(x + (size_t)s * 64 + g * 4);
  float* o = agg + (size_t)d * 64 + g * 4;
  atomicAdd(o + 0, v.x);
  atomicAdd(o + 1, v.y);
  atomicAdd(o + 2, v.z);
  atomicAdd(o + 3, v.w);
}

// agg /= max(deg,1), float4 per thread
__global__ void mean_div4(float4* __restrict__ agg, const float* __restrict__ deg, size_t nq) {
  size_t i = (size_t)blockIdx.x * blockDim.x + threadIdx.x;
  if (i >= nq) return;
  float inv = 1.0f / fmaxf(deg[i >> 4], 1.0f);   // 16 float4 groups per node
  float4 v = agg[i];
  v.x *= inv; v.y *= inv; v.z *= inv; v.w *= inv;
  agg[i] = v;
}

// ------------------------- weight fragment packing -------------------------
// Pack a 64x64 fp32 weight (row-major [K,N]) into bf16 WMMA B-fragments:
// P[((ct*2 + kbi)*32 + lane)*16 + e] ; per CDNA5 ISA 7.12.2 16-bit layout:
//   lanes 0-15 -> K base = kbi*32,   lanes 16-31 -> +8 ; N = ct*16 + (lane&15)
//   elements 0..7 -> K = k0..k0+7 ; elements 8..15 -> K = k0+16..k0+23
#define WPACK_ELEMS 4096  // 4 coltiles * 2 ksteps * 32 lanes * 16 elems

__global__ void pack_w(const float* __restrict__ W, __bf16* __restrict__ P) {
  int t = blockIdx.x * blockDim.x + threadIdx.x;
  if (t >= WPACK_ELEMS) return;
  int e    = t & 15;
  int lane = (t >> 4) & 31;
  int kbi  = (t >> 9) & 1;
  int ct   = (t >> 10) & 3;
  int n  = ct * 16 + (lane & 15);
  int k0 = kbi * 32 + ((lane >> 4) << 3);
  int k  = k0 + (e & 7) + ((e >> 3) << 4);
  P[t] = (__bf16)W[k * 64 + n];
}

// ------------------------------ WMMA GEMM ----------------------------------
// C[nrows,64] = A1 @ W1 (+ A2 @ W2) + bias ; A fp32 in memory (inline cvt),
// W pre-packed bf16 fragments. One wave -> 16x16 tile; 8 waves -> 32x64.

__device__ inline v16bf load_a_frag(const float* __restrict__ A, int row0, int kb, int nrows) {
  int lane = threadIdx.x & 31;
  int r = row0 + (lane & 15);
  if (r >= nrows) r = nrows - 1;                 // clamp (stores are guarded)
  int k0 = kb + ((lane >> 4) << 3);              // lanes 0-15: K base, 16-31: +8
  const float* p = A + (size_t)r * 64 + k0;
  v16bf a;
#pragma unroll
  for (int j = 0; j < 8; ++j) {
    a[j]     = (__bf16)p[j];                     // K = k0..k0+7
    a[8 + j] = (__bf16)p[16 + j];                // K = k0+16..k0+23
  }
  return a;
}

__device__ inline v16bf load_b_packed(const __bf16* __restrict__ P, int ct, int kbi) {
  int lane = threadIdx.x & 31;
  return *(const v16bf*)(P + ((((ct * 2 + kbi) * 32) + lane) << 4));
}

template <bool DUAL>
__global__ void gemm64(const float* __restrict__ A1, const __bf16* __restrict__ P1,
                       const float* __restrict__ A2, const __bf16* __restrict__ P2,
                       const float* __restrict__ bias, float* __restrict__ C, int nrows) {
  int wave = threadIdx.x >> 5;                   // 8 waves per block
  int lane = threadIdx.x & 31;
  int row0 = blockIdx.x * 32 + (wave >> 2) * 16;
  int ct   = wave & 3;
  int col0 = ct * 16;
  if (row0 >= nrows) return;                     // wave-uniform: EXEC all-1 for WMMA

  v8f acc = {};
#pragma unroll
  for (int kbi = 0; kbi < 2; ++kbi) {
    v16bf a = load_a_frag(A1, row0, kbi * 32, nrows);
    v16bf b = load_b_packed(P1, ct, kbi);
    acc = __builtin_amdgcn_wmma_f32_16x16x32_bf16(false, a, false, b, (short)0, acc, false, false);
  }
  if (DUAL) {
#pragma unroll
    for (int kbi = 0; kbi < 2; ++kbi) {
      v16bf a = load_a_frag(A2, row0, kbi * 32, nrows);
      v16bf b = load_b_packed(P2, ct, kbi);
      acc = __builtin_amdgcn_wmma_f32_16x16x32_bf16(false, a, false, b, (short)0, acc, false, false);
    }
  }

  int n  = col0 + (lane & 15);
  int mh = (lane >> 4) * 8;                      // element r -> row r / r+8
  float bv = bias ? bias[n] : 0.0f;
#pragma unroll
  for (int r = 0; r < 8; ++r) {
    int row = row0 + mh + r;
    if (row < nrows) C[(size_t)row * 64 + n] = acc[r] + bv;
  }
}

// ------------------------------ GAT kernels --------------------------------

__global__ void att_kernel(const float* __restrict__ h, const float* __restrict__ a_src,
                           const float* __restrict__ a_dst, float* __restrict__ asrc,
                           float* __restrict__ adst, int Nn, int H, int F) {
  int t = blockIdx.x * blockDim.x + threadIdx.x;
  if (t >= Nn * H) return;
  int n = t / H, hd = t % H;
  const float* hp = h + (size_t)n * 64 + hd * F;
  const float* as = a_src + hd * F;
  const float* ad = a_dst + hd * F;
  float s0 = 0.f, s1 = 0.f;
  for (int f = 0; f < F; ++f) { float v = hp[f]; s0 += v * as[f]; s1 += v * ad[f]; }
  asrc[t] = s0;
  adst[t] = s1;
}

__device__ inline float lrelu(float v) { return v > 0.f ? v : 0.2f * v; }

__device__ inline void atomicMaxF(float* addr, float v) {
  if (v >= 0.f) atomicMax((int*)addr, __float_as_int(v));
  else          atomicMin((unsigned int*)addr, __float_as_uint(v));
}

// one thread per edge; contiguous per-head runs of asrc[s,:] / adst[d,:]
__global__ void edge_max(const int* __restrict__ src, const int* __restrict__ dst,
                         const float* __restrict__ asrc, const float* __restrict__ adst,
                         float* __restrict__ emax, int E, int H) {
  int e = blockIdx.x * blockDim.x + threadIdx.x;
  if (e >= E) return;
  int s = src[e], d = dst[e];
  const float* as = asrc + (size_t)s * H;
  const float* ad = adst + (size_t)d * H;
  float* em = emax + (size_t)d * H;
  for (int hd = 0; hd < H; ++hd)
    atomicMaxF(&em[hd], lrelu(as[hd] + ad[hd]));
}

__global__ void edge_denom(const int* __restrict__ src, const int* __restrict__ dst,
                           const float* __restrict__ asrc, const float* __restrict__ adst,
                           const float* __restrict__ emax, float* __restrict__ denom,
                           int E, int H) {
  int e = blockIdx.x * blockDim.x + threadIdx.x;
  if (e >= E) return;
  int s = src[e], d = dst[e];
  const float* as = asrc + (size_t)s * H;
  const float* ad = adst + (size_t)d * H;
  const float* em = emax + (size_t)d * H;
  float* dn = denom + (size_t)d * H;
  for (int hd = 0; hd < H; ++hd)
    atomicAdd(&dn[hd], expf(lrelu(as[hd] + ad[hd]) - em[hd]));
}

// gout[dst, f] += alpha(dst,head) * h[src, f] ; 4 threads/edge, 16 feats each
__global__ void edge_aggr(const int* __restrict__ src, const int* __restrict__ dst,
                          const float* __restrict__ asrc, const float* __restrict__ adst,
                          const float* __restrict__ emax, const float* __restrict__ denom,
                          const float* __restrict__ h, float* __restrict__ gout,
                          int E, int H, int F) {
  size_t t = (size_t)blockIdx.x * blockDim.x + threadIdx.x;
  int e = (int)(t >> 2);
  if (e >= E) return;
  int c = (int)(t & 3);
  int s = src[e], d = dst[e];
  int f0 = c * 16;
  int curh = -1;
  float alpha = 0.f;
#pragma unroll
  for (int j = 0; j < 16; ++j) {
    int f = f0 + j;
    int hd = f / F;
    if (hd != curh) {
      curh = hd;
      float ev = lrelu(asrc[(size_t)s * H + hd] + adst[(size_t)d * H + hd]);
      alpha = expf(ev - emax[(size_t)d * H + hd]) / denom[(size_t)d * H + hd];
    }
    atomicAdd(&gout[(size_t)d * 64 + f], alpha * h[(size_t)s * 64 + f]);
  }
}

// x_out = x_in + silu(relu(s)) + silu(relu(g + gb)) ; float4 per thread
__device__ inline float relu_silu(float v) {
  float u = v > 0.f ? v : 0.f;
  return u / (1.f + expf(-u));
}

__global__ void combine4(const float4* __restrict__ xin, const float4* __restrict__ s,
                         const float4* __restrict__ g, const float* __restrict__ gb,
                         float4* __restrict__ xout, size_t nq) {
  size_t i = (size_t)blockIdx.x * blockDim.x + threadIdx.x;
  if (i >= nq) return;
  float4 xv = xin[i], sv = s[i], gv = g[i];
  int gbase = (int)(i & 15) * 4;                 // 16 float4 groups per 64-col row
  float4 o;
  o.x = xv.x + relu_silu(sv.x) + relu_silu(gv.x + gb[gbase + 0]);
  o.y = xv.y + relu_silu(sv.y) + relu_silu(gv.y + gb[gbase + 1]);
  o.z = xv.z + relu_silu(sv.z) + relu_silu(gv.z + gb[gbase + 2]);
  o.w = xv.w + relu_silu(sv.w) + relu_silu(gv.w + gb[gbase + 3]);
  xout[i] = o;
}

// ------------------------------ host-side layer ----------------------------

static void launch_layer(const float* xin, float* xout,
                         const int* src, const int* dst, int N, int E,
                         const __bf16* pWl, const __bf16* pWr, const float* sb,
                         const __bf16* pGW, const float* gas, const float* gad, const float* gb,
                         int H, int F,
                         float* deg, float* agg, float* sbuf, float* hbuf, float* gout,
                         float* asrc, float* adst, float* emax, float* denom,
                         hipStream_t stream) {
  size_t nd = (size_t)N * 64;
  size_t nh = (size_t)N * H;

  init_f32<<<gridFor(nd), TPB, 0, stream>>>(agg, 0.0f, nd);
  init_f32<<<gridFor(nd), TPB, 0, stream>>>(gout, 0.0f, nd);
  init_f32<<<gridFor(nh), TPB, 0, stream>>>(emax, -3.0e38f, nh);
  init_f32<<<gridFor(nh), TPB, 0, stream>>>(denom, 0.0f, nh);

  // SAGE: mean aggregate then dual GEMM
  sage_agg<<<gridFor((size_t)E * 16), TPB, 0, stream>>>(xin, src, dst, agg, E);
  mean_div4<<<gridFor(nd / 4), TPB, 0, stream>>>((float4*)agg, deg, nd / 4);
  gemm64<true><<<dim3((N + 31) / 32), TPB, 0, stream>>>(agg, pWl, xin, pWr, sb, sbuf, N);

  // GAT: projection + attention + softmax-weighted aggregation
  gemm64<false><<<dim3((N + 31) / 32), TPB, 0, stream>>>(xin, pGW, nullptr, nullptr, nullptr, hbuf, N);
  att_kernel<<<gridFor((size_t)N * H), TPB, 0, stream>>>(hbuf, gas, gad, asrc, adst, N, H, F);
  edge_max<<<gridFor((size_t)E), TPB, 0, stream>>>(src, dst, asrc, adst, emax, E, H);
  edge_denom<<<gridFor((size_t)E), TPB, 0, stream>>>(src, dst, asrc, adst, emax, denom, E, H);
  edge_aggr<<<gridFor((size_t)E * 4), TPB, 0, stream>>>(src, dst, asrc, adst, emax, denom, hbuf, gout, E, H, F);

  combine4<<<gridFor(nd / 4), TPB, 0, stream>>>((const float4*)xin, (const float4*)sbuf,
                                                (const float4*)gout, gb, (float4*)xout, nd / 4);
}

// ------------------------------ entry point --------------------------------

extern "C" void kernel_launch(void* const* d_in, const int* in_sizes, int n_in,
                              void* d_out, int out_size, void* d_ws, size_t ws_size,
                              hipStream_t stream) {
  const float* x    = (const float*)d_in[0];
  const int*   ei   = (const int*)d_in[1];     // int32 (JAX x64 disabled)
  const float* s1Wl = (const float*)d_in[2];
  const float* s1Wr = (const float*)d_in[3];
  const float* s1b  = (const float*)d_in[4];
  const float* s2Wl = (const float*)d_in[5];
  const float* s2Wr = (const float*)d_in[6];
  const float* s2b  = (const float*)d_in[7];
  const float* g1W  = (const float*)d_in[8];
  const float* g1as = (const float*)d_in[9];
  const float* g1ad = (const float*)d_in[10];
  const float* g1b  = (const float*)d_in[11];
  const float* g2W  = (const float*)d_in[12];
  const float* g2as = (const float*)d_in[13];
  const float* g2ad = (const float*)d_in[14];
  const float* g2b  = (const float*)d_in[15];

  int N = in_sizes[0] / 64;
  int E = in_sizes[1] / 2;
  const int* src = ei;
  const int* dst = ei + E;
  float* out = (float*)d_out;

  // workspace carve: [ packed weights (6 x 4096 bf16) | fp32 scratch ]
  __bf16* pw   = (__bf16*)d_ws;
  __bf16* p1Wl = pw + 0 * WPACK_ELEMS;
  __bf16* p1Wr = pw + 1 * WPACK_ELEMS;
  __bf16* p1GW = pw + 2 * WPACK_ELEMS;
  __bf16* p2Wl = pw + 3 * WPACK_ELEMS;
  __bf16* p2Wr = pw + 4 * WPACK_ELEMS;
  __bf16* p2GW = pw + 5 * WPACK_ELEMS;

  size_t nd = (size_t)N * 64;
  float* ws   = (float*)(pw + 6 * WPACK_ELEMS);  // 48KB in, still well aligned
  float* deg  = ws;  ws += N;
  float* agg  = ws;  ws += nd;
  float* sbuf = ws;  ws += nd;
  float* hbuf = ws;  ws += nd;
  float* gout = ws;  ws += nd;
  float* asrc = ws;  ws += (size_t)N * 8;
  float* adst = ws;  ws += (size_t)N * 8;
  float* emax = ws;  ws += (size_t)N * 8;
  float* denom= ws;  ws += (size_t)N * 8;

  // pack all six 64x64 weights into WMMA B-fragment bf16 layout (once per call)
  pack_w<<<gridFor(WPACK_ELEMS), TPB, 0, stream>>>(s1Wl, p1Wl);
  pack_w<<<gridFor(WPACK_ELEMS), TPB, 0, stream>>>(s1Wr, p1Wr);
  pack_w<<<gridFor(WPACK_ELEMS), TPB, 0, stream>>>(g1W,  p1GW);
  pack_w<<<gridFor(WPACK_ELEMS), TPB, 0, stream>>>(s2Wl, p2Wl);
  pack_w<<<gridFor(WPACK_ELEMS), TPB, 0, stream>>>(s2Wr, p2Wr);
  pack_w<<<gridFor(WPACK_ELEMS), TPB, 0, stream>>>(g2W,  p2GW);

  // in-degree (shared by both layers)
  init_f32<<<gridFor((size_t)N), TPB, 0, stream>>>(deg, 0.0f, (size_t)N);
  deg_kernel<<<gridFor((size_t)E), TPB, 0, stream>>>(dst, deg, E);

  // layer 1: x -> out   (GAT: 8 heads x 8, concat)
  launch_layer(x, out, src, dst, N, E, p1Wl, p1Wr, s1b, p1GW, g1as, g1ad, g1b,
               8, 8, deg, agg, sbuf, hbuf, gout, asrc, adst, emax, denom, stream);

  // layer 2: out -> out (GAT: 1 head x 64, mean == identity)
  launch_layer(out, out, src, dst, N, E, p2Wl, p2Wr, s2b, p2GW, g2as, g2ad, g2b,
               1, 64, deg, agg, sbuf, hbuf, gout, asrc, adst, emax, denom, stream);
}
